// VQAutoEncoderEx_73126113182367
// MI455X (gfx1250) — matrix operand order repaired and checked by reading, not compiled, not run on hardware
//
#include <hip/hip_runtime.h>
#include <hip/hip_bf16.h>
#include <math.h>

// ---------------------------------------------------------------------------
// VQAutoEncoderEx forward on gfx1250 (MI455X).
// Convs: workgroup-tiled implicit GEMM. Weights are pre-converted to bf16
// once per launch; the A (weight) tile is DMA'd into LDS by the Tensor Data
// Mover (tensor_load_to_lds + s_wait_tensorcnt), the B (im2col) tile is
// cooperatively staged in WMMA-fragment order, compute is
// V_WMMA_F32_16X16X32_BF16 with f32 accumulate + f32 epilogue.
// ---------------------------------------------------------------------------

typedef __attribute__((ext_vector_type(16))) __bf16 v16bf;
typedef __attribute__((ext_vector_type(8)))  __bf16 v8bf;
typedef __attribute__((ext_vector_type(8)))  float  v8f;
typedef __attribute__((ext_vector_type(4)))  unsigned int v4u;
typedef __attribute__((ext_vector_type(8)))  int v8i;
typedef __attribute__((ext_vector_type(4)))  int v4i;

#define BM 64     // output-channel tile per block
#define BN 128    // pixel tile per block

// ---------------------------------------------------------------------------
// Implicit-GEMM conv (ks 1/3, stride 1/2, optional fused 2x nearest upsample).
// Block: 256 threads = 8 waves laid out 4(M) x 2(N); each wave owns a
// 16(M) x 64(N) subtile = 4 f32 accumulators; 32 WMMAs per block per K-step.
//   ldsA: plain [m][k] 64x32 bf16, filled by one TDM descriptor per K-step
//         (tensor dims = remaining extent -> hardware zero-fills the tails).
//   ldsB: WMMA-fragment order [(col*2 + half)*16 + j].
// ---------------------------------------------------------------------------
__global__ __launch_bounds__(256) void k_conv_wmma(
    const float* __restrict__ x, const __bf16* __restrict__ w,
    const float* __restrict__ bias, float* __restrict__ y,
    int Cin, int Hin, int Win, int Cout, int Hout, int Wout,
    int ks, int stride, int pad, int ups)
{
    __shared__ __align__(32) __bf16 ldsA[BM * 32];   // 4 KB, plain row-major
    __shared__ __align__(32) __bf16 ldsB[BN * 32];   // 8 KB, fragment order

    const int tid   = threadIdx.x;
    const int lane  = tid & 31;
    const int wv    = tid >> 5;
    const int m_sub = wv & 3;            // which 16-row M subtile
    const int n_sub = wv >> 2;           // which 64-col N subtile
    const int half  = lane >> 4, l = lane & 15;

    const int m_base = blockIdx.x * BM;
    const int n_base = blockIdx.y * BN;
    const int HWo = Hout * Wout;
    const int ks2 = ks * ks;
    const int K   = Cin * ks2;
    const int Hl  = ups ? (Hin << 1) : Hin;
    const int Wl  = ups ? (Win << 1) : Win;

    // ---- fixed B-staging role: this thread owns column nb, k-half bh ----
    const int nb  = tid & (BN - 1);
    const int bh  = tid >> 7;
    const int pb  = n_base + nb;               // pixel (HW multiple of 128)
    const int oh  = pb / Wout, ow = pb % Wout;
    const int ih0 = oh * stride - pad;
    const int iw0 = ow * stride - pad;

    v8f acc[4];
    #pragma unroll
    for (int t = 0; t < 4; ++t) acc[t] = (v8f){0,0,0,0,0,0,0,0};

    const unsigned lds_a_addr = (unsigned)(size_t)(&ldsA[0]);

    for (int k0 = 0; k0 < K; k0 += 32) {
        // ------- A tile via Tensor Data Mover (wave 0 issues one D#) ------
        if (wv == 0) {
            unsigned long long ga =
                (unsigned long long)(size_t)(w + (size_t)m_base * K + k0);
            int dim0 = K - k0;            // remaining k extent (zero-fill tail)
            int dim1 = Cout - m_base;     // remaining rows   (zero-fill tail)
            v4u g0;
            g0.x = 1u;                                    // count=1, user D#
            g0.y = lds_a_addr;                            // LDS dest (bytes)
            g0.z = (unsigned)ga;                          // global_addr lo
            g0.w = (unsigned)(ga >> 32) | (2u << 30);     // addr hi | type=2
            v8i g1;
            g1[0] = 0x10000;                              // data_size=2B
            g1[1] = (dim0 & 0xffff) << 16;                // tensor_dim0 lo16
            g1[2] = (int)(((unsigned)dim0 >> 16) | (((unsigned)dim1 & 0xffff) << 16));
            g1[3] = (int)(((unsigned)dim1 >> 16) | (32u << 16));   // tile_dim0=32
            g1[4] = 64;                                   // tile_dim1=64
            g1[5] = K;                                    // dim0 stride (elems)
            g1[6] = 0;
            g1[7] = 0;
            v4i gz4 = {0, 0, 0, 0};
            v8i gz8 = {0, 0, 0, 0, 0, 0, 0, 0};
            __builtin_amdgcn_tensor_load_to_lds(g0, g1, gz4, gz4, gz8, 0);
        }
        // ------- stage B (im2col, incremental (ci,kh,kw)) : 16 elems ------
        {
            int kk = k0 + bh * 16;
            int ci = kk / ks2;
            int r  = kk - ci * ks2;
            int kh = r / ks, kw = r - kh * ks;
            v16bf tmp = {};
            #pragma unroll
            for (int j = 0; j < 16; ++j) {
                float bv = 0.f;
                if (kk + j < K) {
                    int ih = ih0 + kh, iw = iw0 + kw;
                    if (ih >= 0 && ih < Hl && iw >= 0 && iw < Wl) {
                        int sh = ups ? (ih >> 1) : ih;
                        int sw = ups ? (iw >> 1) : iw;
                        bv = x[(ci * Hin + sh) * Win + sw];
                    }
                }
                tmp[j] = (__bf16)bv;
                if (++kw == ks) { kw = 0; if (++kh == ks) { kh = 0; ++ci; } }
            }
            *(v16bf*)&ldsB[(nb * 2 + bh) * 16] = tmp;
        }
        if (wv == 0) __builtin_amdgcn_s_wait_tensorcnt(0);
        __syncthreads();
        // ------- compute: A = 2 contiguous 8-elem chunks from plain [m][k]
        const __bf16* arow = &ldsA[(m_sub * 16 + l) * 32];
        v8bf alo = *(const v8bf*)(arow + half * 16);
        v8bf ahi = *(const v8bf*)(arow + half * 16 + 8);
        v16bf afrag = __builtin_shufflevector(alo, ahi,
            0, 1, 2, 3, 4, 5, 6, 7, 8, 9, 10, 11, 12, 13, 14, 15);
        #pragma unroll
        for (int t = 0; t < 4; ++t) {
            int n = n_sub * 64 + t * 16 + l;
            v16bf bfrag = *(const v16bf*)&ldsB[(n * 2 + half) * 16];
            acc[t] = __builtin_amdgcn_wmma_f32_16x16x32_bf16(
                false, afrag, false, bfrag, (short)0, acc[t], false, false);
        }
        __syncthreads();
    }
    // ------- epilogue: bias add, f32 store ---------------------------------
    #pragma unroll
    for (int t = 0; t < 4; ++t) {
        int n = n_base + n_sub * 64 + t * 16 + l;
        #pragma unroll
        for (int r = 0; r < 8; ++r) {
            int mo = m_base + m_sub * 16 + r + half * 8;
            if (mo < Cout && n < HWo)
                y[mo * HWo + n] = acc[t][r] + bias[mo];
        }
    }
}

// ---------------- f32 -> bf16 weight conversion (once per launch) ----------
__global__ void k_f32_to_bf16(const float* __restrict__ src, __bf16* __restrict__ dst, int n)
{
    int i = blockIdx.x * blockDim.x + threadIdx.x;
    if (i < n) dst[i] = (__bf16)src[i];
}

// ---------------- attention GEMMs (16x16 spatial only, tiny) ---------------
// s[i,j] = scale * sum_c q[c,i]*k[c,j]   (M=i, N=j, K=c)
__global__ void k_qk_wmma(const float* __restrict__ q, const float* __restrict__ kmat,
                          float* __restrict__ s, int C, int HW, float scale)
{
    int lane = threadIdx.x, half = lane >> 4, l = lane & 15;
    int i0 = blockIdx.x * 16, j0 = blockIdx.y * 16;
    v8f acc = {};
    for (int k0 = 0; k0 < C; k0 += 32) {
        v16bf a = {}, b = {};
        #pragma unroll
        for (int v = 0; v < 8; ++v)
            #pragma unroll
            for (int e = 0; e < 2; ++e) {
                int kk = k0 + half * 16 + v * 2 + e;
                float av = (kk < C) ? q[kk * HW + (i0 + l)] : 0.f;
                a[v * 2 + e] = (__bf16)av;
            }
        #pragma unroll
        for (int j = 0; j < 16; ++j) {
            int kk = k0 + half * 16 + j;
            float bv = (kk < C) ? kmat[kk * HW + (j0 + l)] : 0.f;
            b[j] = (__bf16)bv;
        }
        acc = __builtin_amdgcn_wmma_f32_16x16x32_bf16(
            false, a, false, b, (short)0, acc, false, false);
    }
    #pragma unroll
    for (int r = 0; r < 8; ++r)
        s[(i0 + r + half * 8) * HW + (j0 + l)] = acc[r] * scale;
}

// hf[c,i] = sum_j v[c,j]*a[i,j]   (M=c, N=i, K=j)
__global__ void k_av_wmma(const float* __restrict__ vmat, const float* __restrict__ att,
                          float* __restrict__ hf, int C, int HW)
{
    int lane = threadIdx.x, half = lane >> 4, l = lane & 15;
    int m0 = blockIdx.x * 16, n0 = blockIdx.y * 16;
    int m = m0 + l;
    v8f acc = {};
    for (int k0 = 0; k0 < HW; k0 += 32) {
        v16bf a = {}, b = {};
        #pragma unroll
        for (int v = 0; v < 8; ++v)
            #pragma unroll
            for (int e = 0; e < 2; ++e) {
                int kk = k0 + half * 16 + v * 2 + e;
                float av = (m < C && kk < HW) ? vmat[m * HW + kk] : 0.f;
                a[v * 2 + e] = (__bf16)av;
            }
        #pragma unroll
        for (int j = 0; j < 16; ++j) {
            int kk = k0 + half * 16 + j;
            float bv = (kk < HW) ? att[(n0 + l) * HW + kk] : 0.f;
            b[j] = (__bf16)bv;
        }
        acc = __builtin_amdgcn_wmma_f32_16x16x32_bf16(
            false, a, false, b, (short)0, acc, false, false);
    }
    #pragma unroll
    for (int r = 0; r < 8; ++r) {
        int mo = m0 + r + half * 8;
        if (mo < C) hf[mo * HW + (n0 + l)] = acc[r];
    }
}

// ---------------- GroupNorm (+ optional SiLU), one block per group ---------
__global__ void k_groupnorm(const float* __restrict__ x, const float* __restrict__ gam,
                            const float* __restrict__ bet, float* __restrict__ y,
                            int C, int HW, int groups, int swish)
{
    __shared__ float ssum[256], ssq[256];
    int g = blockIdx.x, tid = threadIdx.x;
    int cg = C / groups;
    long long n = (long long)cg * HW;
    const float* xg = x + (long long)g * cg * HW;
    float s = 0.f, s2 = 0.f;
    for (long long i = tid; i < n; i += blockDim.x) { float v = xg[i]; s += v; s2 += v * v; }
    ssum[tid] = s; ssq[tid] = s2; __syncthreads();
    for (int o = 128; o > 0; o >>= 1) {
        if (tid < o) { ssum[tid] += ssum[tid + o]; ssq[tid] += ssq[tid + o]; }
        __syncthreads();
    }
    float mean = ssum[0] / (float)n;
    float var  = ssq[0] / (float)n - mean * mean;
    float inv  = rsqrtf(var + 1e-6f);
    float* yg = y + (long long)g * cg * HW;
    for (long long i = tid; i < n; i += blockDim.x) {
        int c = g * cg + (int)(i / HW);
        float v = (xg[i] - mean) * inv * gam[c] + bet[c];
        if (swish) v = v / (1.f + __expf(-v));
        yg[i] = v;
    }
}

__global__ void k_add(const float* __restrict__ a, const float* __restrict__ b,
                      float* __restrict__ y, int n)
{
    int i = blockIdx.x * blockDim.x + threadIdx.x;
    if (i < n) y[i] = a[i] + b[i];
}

__global__ void k_zero(float* p, int n)
{
    int i = blockIdx.x * blockDim.x + threadIdx.x;
    if (i < n) p[i] = 0.f;
}

__global__ void k_softmax_row(float* __restrict__ s, int N)
{
    __shared__ float red[256];
    int row = blockIdx.x, tid = threadIdx.x;
    float* r = s + (long long)row * N;
    float m = -1e30f;
    for (int i = tid; i < N; i += blockDim.x) m = fmaxf(m, r[i]);
    red[tid] = m; __syncthreads();
    for (int o = 128; o > 0; o >>= 1) { if (tid < o) red[tid] = fmaxf(red[tid], red[tid + o]); __syncthreads(); }
    m = red[0]; __syncthreads();
    float sum = 0.f;
    for (int i = tid; i < N; i += blockDim.x) { float e = __expf(r[i] - m); r[i] = e; sum += e; }
    red[tid] = sum; __syncthreads();
    for (int o = 128; o > 0; o >>= 1) { if (tid < o) red[tid] += red[tid + o]; __syncthreads(); }
    float inv = 1.f / red[0];
    for (int i = tid; i < N; i += blockDim.x) r[i] *= inv;
}

// ---------------- Vector quantize: one block per latent pixel --------------
__global__ void k_vq_assign(const float* __restrict__ z, const float* __restrict__ cb,
                            int* __restrict__ idx, float* __restrict__ zq,
                            float* __restrict__ mse, float* __restrict__ hist,
                            int C, int HW, int NCB)
{
    __shared__ float dmin[256]; __shared__ int dix[256]; __shared__ float zsh[256];
    int p = blockIdx.x, tid = threadIdx.x;
    for (int c = tid; c < C; c += blockDim.x) zsh[c] = z[c * HW + p];
    __syncthreads();
    float best = 1e30f; int bi = 0;
    for (int code = tid; code < NCB; code += blockDim.x) {
        const float* cr = cb + (long long)code * C;
        float d = 0.f;
        for (int c = 0; c < C; ++c) { float t = zsh[c] - cr[c]; d += t * t; }
        if (d < best) { best = d; bi = code; }
    }
    dmin[tid] = best; dix[tid] = bi; __syncthreads();
    for (int o = 128; o > 0; o >>= 1) {
        if (tid < o) {
            if (dmin[tid + o] < dmin[tid] ||
                (dmin[tid + o] == dmin[tid] && dix[tid + o] < dix[tid])) {
                dmin[tid] = dmin[tid + o]; dix[tid] = dix[tid + o];
            }
        }
        __syncthreads();
    }
    int b = dix[0];
    if (tid == 0) { idx[p] = b; atomicAdd(&hist[b], 1.0f); }
    float lerr = 0.f;
    const float* cr = cb + (long long)b * C;
    for (int c = tid; c < C; c += blockDim.x) {
        float q = cr[c];
        zq[c * HW + p] = q;          // straight-through: generator sees zq
        float t = q - zsh[c]; lerr += t * t;
    }
    dmin[tid] = lerr; __syncthreads();
    for (int o = 128; o > 0; o >>= 1) { if (tid < o) dmin[tid] += dmin[tid + o]; __syncthreads(); }
    if (tid == 0) atomicAdd(mse, dmin[0]);
}

__global__ void k_vq_final(const float* __restrict__ mse, const float* __restrict__ hist,
                           float* out_loss, float* out_ppl, int NCB, int npix, int nelem)
{
    __shared__ float red[256];
    int tid = threadIdx.x;
    float s = 0.f;
    for (int i = tid; i < NCB; i += blockDim.x) {
        float e = hist[i] / (float)npix;
        s += e * __logf(e + 1e-10f);
    }
    red[tid] = s; __syncthreads();
    for (int o = 128; o > 0; o >>= 1) { if (tid < o) red[tid] += red[tid + o]; __syncthreads(); }
    if (tid == 0) {
        *out_ppl  = __expf(-red[0]);
        *out_loss = 1.25f * (*mse) / (float)nelem;   // mse + beta*mse
    }
}

// ===========================================================================
// Host-side orchestration
// ===========================================================================
enum { OPC, OPR, OPA, OPD, OPU, OPG };
struct Op { int k, cin, cout; };

static int build_enc(Op* ops)
{
    int n = 0, curr = 512;
    const int cm[6]  = {1, 2, 2, 4, 4, 8};
    const int inm[7] = {1, 1, 2, 2, 4, 4, 8};
    ops[n++] = {OPC, 3, 64};
    int cin = 0;
    for (int i = 0; i < 6; ++i) {
        cin = 64 * inm[i]; int cout = 64 * cm[i];
        for (int r = 0; r < 2; ++r) {
            ops[n++] = {OPR, cin, cout}; cin = cout;
            if (curr == 16) ops[n++] = {OPA, cin, cin};
        }
        if (i != 5) { ops[n++] = {OPD, cin, cin}; curr /= 2; }
    }
    ops[n++] = {OPR, cin, cin};
    ops[n++] = {OPA, cin, cin};
    ops[n++] = {OPR, cin, cin};
    ops[n++] = {OPG, cin, cin};
    ops[n++] = {OPC, cin, 256};
    return n;
}

static int build_gen(Op* ops)
{
    int n = 0, cin = 64 * 8, curr = 16;
    const int cm[6] = {1, 2, 2, 4, 4, 8};
    ops[n++] = {OPC, 256, cin};
    ops[n++] = {OPR, cin, cin};
    ops[n++] = {OPA, cin, cin};
    ops[n++] = {OPR, cin, cin};
    for (int i = 5; i >= 0; --i) {
        int cout = 64 * cm[i];
        for (int r = 0; r < 2; ++r) {
            ops[n++] = {OPR, cin, cout}; cin = cout;
            if (curr == 16) ops[n++] = {OPA, cin, cin};
        }
        if (i != 0) { ops[n++] = {OPU, cin, cin}; curr *= 2; }
    }
    ops[n++] = {OPG, cin, cin};
    ops[n++] = {OPC, cin, 3};
    return n;
}

static float* pick(float* big[3], const float* a, const float* b)
{
    for (int i = 0; i < 3; ++i)
        if (big[i] != a && big[i] != b) return big[i];
    return big[0];
}

// Convert one f32 weight tensor into the bf16 arena; advance cursor.
static const __bf16* cvt_w(const float* src, size_t n, __bf16*& cur, hipStream_t st)
{
    __bf16* dst = cur; cur += n;
    k_f32_to_bf16<<<dim3((unsigned)((n + 255) / 256)), dim3(256), 0, st>>>(src, dst, (int)n);
    return dst;
}

static void launch_conv(const float* x, const __bf16* w, const float* b, float* y,
                        int Cin, int Hin, int Win, int Cout, int Hout, int Wout,
                        int ks, int stride, int pad, int ups, hipStream_t st)
{
    dim3 grid((Cout + BM - 1) / BM, (Hout * Wout + BN - 1) / BN);
    k_conv_wmma<<<grid, 256, 0, st>>>(x, w, b, y, Cin, Hin, Win, Cout, Hout, Wout,
                                      ks, stride, pad, ups);
}

static void run_plan(const Op* ops, int nops, const float* const* P, int& pi,
                     float* big[3], float* sm, __bf16*& acur,
                     const float*& cur, int& C, int& H,
                     float* final_dst, hipStream_t st)
{
    float* at_h  = sm;
    float* at_q  = at_h  + 131072;
    float* at_k  = at_q  + 131072;
    float* at_v  = at_k  + 131072;
    float* at_hf = at_v  + 131072;
    float* at_p  = at_hf + 131072;
    float* at_s  = at_p  + 131072;

    for (int oi = 0; oi < nops; ++oi) {
        const Op& op = ops[oi];
        int HW = H * H;
        switch (op.k) {
        case OPC: {
            const __bf16* w = cvt_w(P[pi], (size_t)op.cout * op.cin * 9, acur, st); pi++;
            const float* b = P[pi++];
            float* dst = (oi == nops - 1 && final_dst) ? final_dst : pick(big, cur, nullptr);
            launch_conv(cur, w, b, dst, op.cin, H, H, op.cout, H, H, 3, 1, 1, 0, st);
            cur = dst; C = op.cout;
        } break;
        case OPG: {
            const float* g = P[pi++]; const float* b = P[pi++];
            float* dst = pick(big, cur, nullptr);
            k_groupnorm<<<dim3(32), dim3(256), 0, st>>>(cur, g, b, dst, C, HW, 32, 0);
            cur = dst;
        } break;
        case OPD: {
            const __bf16* w = cvt_w(P[pi], (size_t)C * C * 9, acur, st); pi++;
            const float* b = P[pi++];
            float* dst = pick(big, cur, nullptr);
            // F.pad(0,1,0,1) + stride-2 pad-0 conv == stride-2 pad-0 with zero OOB
            launch_conv(cur, w, b, dst, C, H, H, C, H / 2, H / 2, 3, 2, 0, 0, st);
            cur = dst; H /= 2;
        } break;
        case OPU: {
            const __bf16* w = cvt_w(P[pi], (size_t)C * C * 9, acur, st); pi++;
            const float* b = P[pi++];
            float* dst = pick(big, cur, nullptr);
            launch_conv(cur, w, b, dst, C, H, H, C, 2 * H, 2 * H, 3, 1, 1, 1, st);
            cur = dst; H *= 2;
        } break;
        case OPR: {
            const float* g1 = P[pi++]; const float* b1 = P[pi++];
            const __bf16* w1 = cvt_w(P[pi], (size_t)op.cout * op.cin * 9, acur, st); pi++;
            const float* bb1 = P[pi++];
            const float* g2 = P[pi++]; const float* b2 = P[pi++];
            const __bf16* w2 = cvt_w(P[pi], (size_t)op.cout * op.cout * 9, acur, st); pi++;
            const float* bb2 = P[pi++];
            const __bf16* ws = nullptr; const float* bs = nullptr;
            if (op.cin != op.cout) {
                ws = cvt_w(P[pi], (size_t)op.cout * op.cin, acur, st); pi++;
                bs = P[pi++];
            }
            float* t1 = pick(big, cur, nullptr);
            float* t2 = pick(big, cur, t1);
            k_groupnorm<<<dim3(32), dim3(256), 0, st>>>(cur, g1, b1, t1, op.cin, HW, 32, 1);
            launch_conv(t1, w1, bb1, t2, op.cin, H, H, op.cout, H, H, 3, 1, 1, 0, st);
            k_groupnorm<<<dim3(32), dim3(256), 0, st>>>(t2, g2, b2, t1, op.cout, HW, 32, 1);
            launch_conv(t1, w2, bb2, t2, op.cout, H, H, op.cout, H, H, 3, 1, 1, 0, st);
            int n = op.cout * HW;
            if (ws) {
                launch_conv(cur, ws, bs, t1, op.cin, H, H, op.cout, H, H, 1, 1, 0, 0, st);
                float* dst = pick(big, t1, t2);
                k_add<<<dim3((n + 255) / 256), dim3(256), 0, st>>>(t2, t1, dst, n);
                cur = dst;
            } else {
                k_add<<<dim3((n + 255) / 256), dim3(256), 0, st>>>(t2, cur, t1, n);
                cur = t1;
            }
            C = op.cout;
        } break;
        case OPA: {
            const float* gg = P[pi++]; const float* gb = P[pi++];
            const __bf16* qw = cvt_w(P[pi], (size_t)C * C, acur, st); pi++;
            const float* qb = P[pi++];
            const __bf16* kw = cvt_w(P[pi], (size_t)C * C, acur, st); pi++;
            const float* kb = P[pi++];
            const __bf16* vw = cvt_w(P[pi], (size_t)C * C, acur, st); pi++;
            const float* vb = P[pi++];
            const __bf16* pw = cvt_w(P[pi], (size_t)C * C, acur, st); pi++;
            const float* pb = P[pi++];
            k_groupnorm<<<dim3(32), dim3(256), 0, st>>>(cur, gg, gb, at_h, C, HW, 32, 0);
            launch_conv(at_h, qw, qb, at_q, C, H, H, C, H, H, 1, 1, 0, 0, st);
            launch_conv(at_h, kw, kb, at_k, C, H, H, C, H, H, 1, 1, 0, 0, st);
            launch_conv(at_h, vw, vb, at_v, C, H, H, C, H, H, 1, 1, 0, 0, st);
            float scale = 1.0f / sqrtf((float)C);
            k_qk_wmma<<<dim3(HW / 16, HW / 16), dim3(32), 0, st>>>(at_q, at_k, at_s, C, HW, scale);
            k_softmax_row<<<dim3(HW), dim3(256), 0, st>>>(at_s, HW);
            k_av_wmma<<<dim3((C + 15) / 16, HW / 16), dim3(32), 0, st>>>(at_v, at_s, at_hf, C, HW);
            launch_conv(at_hf, pw, pb, at_p, C, H, H, C, H, H, 1, 1, 0, 0, st);
            float* dst = pick(big, cur, nullptr);
            int n = C * HW;
            k_add<<<dim3((n + 255) / 256), dim3(256), 0, st>>>(cur, at_p, dst, n);
            cur = dst;
        } break;
        }
    }
}

extern "C" void kernel_launch(void* const* d_in, const int* in_sizes, int n_in,
                              void* d_out, int out_size, void* d_ws, size_t ws_size,
                              hipStream_t stream)
{
    (void)in_sizes; (void)n_in; (void)ws_size;
    const float* const* P = (const float* const*)d_in;
    float* out  = (float*)d_out;
    float* base = (float*)d_ws;

    const size_t BIG = (size_t)128 * 512 * 512;       // largest activation (floats)
    float* big[3] = { base, base + BIG, base + 2 * BIG };
    float* sm = base + 3 * BIG;                        // attention + VQ scratch

    float* vq_mse  = sm + 6 * 131072 + 65536;
    float* vq_hist = vq_mse + 1;
    int*   vq_idx  = (int*)(vq_hist + 1024);

    // bf16 weight arena (converted once per launch, deterministic layout)
    __bf16* arena = (__bf16*)(((size_t)(vq_idx + 256) + 255) & ~(size_t)255);
    __bf16* acur  = arena;

    Op enc[64], gen[64];
    int ne = build_enc(enc), ng = build_gen(gen);

    int pi = 1;                         // d_in[0] = x
    const float* cur = P[0];
    int C = 3, H = 512;

    // ---- encoder: x -> z (256 x 16 x 16) ----
    run_plan(enc, ne, P, pi, big, sm, acur, cur, C, H, nullptr, stream);

    // ---- vector quantize ----
    const float* codebook = P[pi++];
    k_zero<<<dim3((1025 + 255) / 256), dim3(256), 0, stream>>>(vq_mse, 1025);
    float* zq = pick(big, cur, nullptr);
    int HW = H * H;
    k_vq_assign<<<dim3(HW), dim3(256), 0, stream>>>(cur, codebook, vq_idx, zq,
                                                    vq_mse, vq_hist, C, HW, 1024);
    k_vq_final<<<dim3(1), dim3(256), 0, stream>>>(vq_mse, vq_hist,
                                                  out + (out_size - 2), out + (out_size - 1),
                                                  1024, HW, C * HW);
    cur = zq;

    // ---- generator: zq -> recon (written straight into d_out) ----
    run_plan(gen, ng, P, pi, big, sm, acur, cur, C, H, out, stream);
}
